// RNN_LSTM_one_step_90907277787355
// MI455X (gfx1250) — matrix operand order repaired and checked by reading, not compiled, not run on hardware
//
#include <hip/hip_runtime.h>

#define B_ 128
#define T_ 512
#define I_ 256
#define H_ 1024
#define C_ 1000
#define NPAD_ 1024  // fc_W padded rows (multiple of 16 >= 1000)

typedef __attribute__((ext_vector_type(16))) _Float16 v16h;
typedef __attribute__((ext_vector_type(8)))  _Float16 v8h;
typedef __attribute__((ext_vector_type(8)))  float    v8f;

// ---------------------------------------------------------------------------
// WMMA fragment loaders (wave32, per CDNA5 ISA 7.12.2 layouts)
// ---------------------------------------------------------------------------

// A-matrix 16x32 f16 (MxK): lane holds row m=lane&15; two contiguous 8-half
// chunks at k = half*8 and k = 16 + half*8 (half = lane>>4).
__device__ __forceinline__ v16h load_a_frag(const _Float16* __restrict__ base, int ld) {
  const int lane = __lane_id();
  const int row  = lane & 15;
  const int half = lane >> 4;
  const _Float16* p = base + row * ld + half * 8;
  v8h c0 = *(const v8h*)(p);        // k = half*8 .. half*8+7
  v8h c1 = *(const v8h*)(p + 16);   // k = 16+half*8 .. +7
  v16h r;
#pragma unroll
  for (int i = 0; i < 8; ++i) { r[i] = c0[i]; r[i + 8] = c1[i]; }
  return r;
}

// B-matrix 32x16 f16 (KxN): lane holds column n=lane&15; two contiguous 8-half
// chunks at k = half*8 and k = 16 + half*8.  Column n of B == row (n0+n) of the
// row-major weight matrix (we compute A x W^T), so no transpose is needed.
__device__ __forceinline__ v16h load_b_frag(const _Float16* __restrict__ wrow_base, int ld) {
  const int lane = __lane_id();
  const int col  = lane & 15;
  const int half = lane >> 4;
  const _Float16* p = wrow_base + col * ld + half * 8;
  v8h c0 = *(const v8h*)(p);
  v8h c1 = *(const v8h*)(p + 16);
  v16h r;
#pragma unroll
  for (int i = 0; i < 8; ++i) { r[i] = c0[i]; r[i + 8] = c1[i]; }
  return r;
}

__device__ __forceinline__ v8f wmma16x16x32(v16h a, v16h b, v8f c) {
  return __builtin_amdgcn_wmma_f32_16x16x32_f16(
      /*neg_a=*/false, a, /*neg_b=*/false, b,
      /*c_mod=*/(short)0, c, /*reuse_a=*/false, /*reuse_b=*/false);
}

// ---------------------------------------------------------------------------
// Software-pipelined, fully unrolled GEMM segment:
//   acc[q][s] += A(m0+16s, 0..LEN) * W(qH+n0, 0..LEN)^T
// LEN is a compile-time constant so every load in the unrolled loop is
// base-pointer + immediate offset (no per-iteration address VALU), and the
// next iteration's 6 fragment loads are issued before the current iteration's
// 8 WMMAs so L2 latency overlaps matrix-core work.
// ---------------------------------------------------------------------------
template <int LEN>
__device__ __forceinline__ void gemm_segment(
    const _Float16* __restrict__ src, int ldA,    // activations [B, ldA]
    const _Float16* __restrict__ W,   int ldW,    // weights [4H, ldW] row-major
    int m0, int n0, v8f (&acc)[4][2])
{
  const _Float16* a0p = src + (size_t)(m0 +  0) * ldA;
  const _Float16* a1p = src + (size_t)(m0 + 16) * ldA;
  const _Float16* b0p = W + (size_t)(0 * H_ + n0) * ldW;
  const _Float16* b1p = W + (size_t)(1 * H_ + n0) * ldW;
  const _Float16* b2p = W + (size_t)(2 * H_ + n0) * ldW;
  const _Float16* b3p = W + (size_t)(3 * H_ + n0) * ldW;

  // Prologue: fragments for k = 0
  v16h a0 = load_a_frag(a0p, ldA);
  v16h a1 = load_a_frag(a1p, ldA);
  v16h b0 = load_b_frag(b0p, ldW);
  v16h b1 = load_b_frag(b1p, ldW);
  v16h b2 = load_b_frag(b2p, ldW);
  v16h b3 = load_b_frag(b3p, ldW);

#pragma unroll
  for (int k = 32; k < LEN; k += 32) {
    // Issue next iteration's loads first (no dependency on current WMMAs);
    // k is a constant => all addresses fold into immediate offsets.
    v16h na0 = load_a_frag(a0p + k, ldA);
    v16h na1 = load_a_frag(a1p + k, ldA);
    v16h nb0 = load_b_frag(b0p + k, ldW);
    v16h nb1 = load_b_frag(b1p + k, ldW);
    v16h nb2 = load_b_frag(b2p + k, ldW);
    v16h nb3 = load_b_frag(b3p + k, ldW);

    acc[0][0] = wmma16x16x32(a0, b0, acc[0][0]);
    acc[0][1] = wmma16x16x32(a1, b0, acc[0][1]);
    acc[1][0] = wmma16x16x32(a0, b1, acc[1][0]);
    acc[1][1] = wmma16x16x32(a1, b1, acc[1][1]);
    acc[2][0] = wmma16x16x32(a0, b2, acc[2][0]);
    acc[2][1] = wmma16x16x32(a1, b2, acc[2][1]);
    acc[3][0] = wmma16x16x32(a0, b3, acc[3][0]);
    acc[3][1] = wmma16x16x32(a1, b3, acc[3][1]);

    a0 = na0; a1 = na1; b0 = nb0; b1 = nb1; b2 = nb2; b3 = nb3;
  }

  // Epilogue: last iteration's WMMAs
  acc[0][0] = wmma16x16x32(a0, b0, acc[0][0]);
  acc[0][1] = wmma16x16x32(a1, b0, acc[0][1]);
  acc[1][0] = wmma16x16x32(a0, b1, acc[1][0]);
  acc[1][1] = wmma16x16x32(a1, b1, acc[1][1]);
  acc[2][0] = wmma16x16x32(a0, b2, acc[2][0]);
  acc[2][1] = wmma16x16x32(a1, b2, acc[2][1]);
  acc[3][0] = wmma16x16x32(a0, b3, acc[3][0]);
  acc[3][1] = wmma16x16x32(a1, b3, acc[3][1]);
}

// ---------------------------------------------------------------------------
// Fast activations (clamped so __expf never overflows)
// ---------------------------------------------------------------------------
__device__ __forceinline__ float fast_sig(float x) {
  x = fminf(fmaxf(x, -30.0f), 30.0f);
  return 1.0f / (1.0f + __expf(-x));
}
__device__ __forceinline__ float fast_tanh(float x) {
  x = fminf(fmaxf(x, -15.0f), 15.0f);
  const float e = __expf(2.0f * x);
  return (e - 1.0f) / (e + 1.0f);
}

// ---------------------------------------------------------------------------
// Pre-pass conversion kernels
// ---------------------------------------------------------------------------

// x [B,T,I] f32 -> x16 [T,B,I] f16 (each timestep's [B,I] tile contiguous)
__global__ void k_cvt_x(const float* __restrict__ x, _Float16* __restrict__ x16, int total) {
  for (int idx = blockIdx.x * blockDim.x + threadIdx.x; idx < total;
       idx += gridDim.x * blockDim.x) {
    const int i  = idx % I_;
    const int bt = idx / I_;
    const int t  = bt % T_;
    const int b  = bt / T_;
    x16[((size_t)t * B_ + b) * I_ + i] = (_Float16)x[idx];
  }
}

__global__ void k_cvt(const float* __restrict__ s, _Float16* __restrict__ d, int n) {
  for (int idx = blockIdx.x * blockDim.x + threadIdx.x; idx < n;
       idx += gridDim.x * blockDim.x)
    d[idx] = (_Float16)s[idx];
}

// fc_W [C,H] f32 -> fc16 [NPAD,H] f16, rows >= C zero-padded
__global__ void k_fc_pad(const float* __restrict__ w, _Float16* __restrict__ d) {
  const int total = NPAD_ * H_;
  for (int idx = blockIdx.x * blockDim.x + threadIdx.x; idx < total;
       idx += gridDim.x * blockDim.x) {
    const int r = idx / H_;
    d[idx] = (r < C_) ? (_Float16)w[idx] : (_Float16)0.0f;
  }
}

// bias = b_ih + b_hh ; h0 = 0 ; c0 = 0
__global__ void k_init(const float* __restrict__ bih, const float* __restrict__ bhh,
                       float* __restrict__ bias, _Float16* __restrict__ h0,
                       float* __restrict__ c0) {
  for (int idx = blockIdx.x * blockDim.x + threadIdx.x; idx < B_ * H_;
       idx += gridDim.x * blockDim.x) {
    h0[idx] = (_Float16)0.0f;
    c0[idx] = 0.0f;
    if (idx < 4 * H_) bias[idx] = bih[idx] + bhh[idx];
  }
}

// ---------------------------------------------------------------------------
// Fused LSTM step: gates = x_t @ Wih^T + h @ Whh^T + bias ; cell update.
//
// Block = 256 threads = 8 waves (2 per SIMD32 for latency hiding); all waves
// share the batch tile m0..m0+31, each wave owns 16 hidden columns and all 4
// gates (8 WMMA accumulators).  Grid: (B/32, H/128) = (4, 8).
// ---------------------------------------------------------------------------
__global__ __launch_bounds__(256) void k_lstm_step(
    const _Float16* __restrict__ xt,    // [B,I]  f16, this timestep
    const _Float16* __restrict__ Wih,   // [4H,I] f16 row-major
    const _Float16* __restrict__ Whh,   // [4H,H] f16 row-major
    const float*    __restrict__ bias,  // [4H]   f32 (b_ih+b_hh)
    const _Float16* __restrict__ hin,   // [B,H]  f16
    float*          __restrict__ cst,   // [B,H]  f32, updated in place
    _Float16*       __restrict__ hout)  // [B,H]  f16
{
  const int wave = threadIdx.x >> 5;
  const int m0 = blockIdx.x * 32;                  // batch tile base (2 sub-tiles)
  const int n0 = blockIdx.y * 128 + wave * 16;     // hidden tile base (per wave)

  v8f acc[4][2];                                   // [gate][m sub-tile]
#pragma unroll
  for (int q = 0; q < 4; ++q)
#pragma unroll
    for (int s = 0; s < 2; ++s) acc[q][s] = (v8f){0,0,0,0,0,0,0,0};

  gemm_segment<I_>(xt,  I_, Wih, I_, m0, n0, acc);  // input projection
  gemm_segment<H_>(hin, H_, Whh, H_, m0, n0, acc);  // recurrent projection

  // Cell update. D layout: element r -> (m = tile_m0 + 8*(lane>=16) + r,
  // n = n0 + (lane&15)).
  const int lane = __lane_id();
  const int n    = n0 + (lane & 15);
  const float bi = bias[0 * H_ + n];
  const float bf = bias[1 * H_ + n];
  const float bg = bias[2 * H_ + n];
  const float bo = bias[3 * H_ + n];
#pragma unroll
  for (int s = 0; s < 2; ++s) {
    const int mbase = m0 + s * 16 + (lane >> 4) * 8;
#pragma unroll
    for (int r = 0; r < 8; ++r) {
      const int m = mbase + r;
      const float gi = fast_sig (acc[0][s][r] + bi);
      const float gf = fast_sig (acc[1][s][r] + bf);
      const float gg = fast_tanh(acc[2][s][r] + bg);
      const float go = fast_sig (acc[3][s][r] + bo);
      const float cn = gf * cst[(size_t)m * H_ + n] + gi * gg;
      cst[(size_t)m * H_ + n]  = cn;
      hout[(size_t)m * H_ + n] = (_Float16)(go * fast_tanh(cn));
    }
  }
}

// ---------------------------------------------------------------------------
// Final FC: out[B,C] = h @ fc_W^T + fc_b   (N padded to 1024, store n < C)
// Grid: (B/32, NPAD/64) = (4, 16), 4 waves/block.
// ---------------------------------------------------------------------------
__global__ __launch_bounds__(128) void k_fc(
    const _Float16* __restrict__ h,    // [B,H] f16
    const _Float16* __restrict__ fcw,  // [NPAD,H] f16
    const float*    __restrict__ fcb,  // [C] f32
    float*          __restrict__ out)  // [B,C] f32
{
  const int wave = threadIdx.x >> 5;
  const int m0 = blockIdx.x * 32;
  const int n0 = blockIdx.y * 64 + wave * 16;

  const _Float16* a0p = h + (size_t)(m0 +  0) * H_;
  const _Float16* a1p = h + (size_t)(m0 + 16) * H_;
  const _Float16* bp  = fcw + (size_t)n0 * H_;

  v8f acc0 = (v8f){0,0,0,0,0,0,0,0};
  v8f acc1 = (v8f){0,0,0,0,0,0,0,0};
#pragma unroll
  for (int k = 0; k < H_; k += 32) {
    const v16h a0 = load_a_frag(a0p + k, H_);
    const v16h a1 = load_a_frag(a1p + k, H_);
    const v16h b  = load_b_frag(bp  + k, H_);
    acc0 = wmma16x16x32(a0, b, acc0);
    acc1 = wmma16x16x32(a1, b, acc1);
  }

  const int lane = __lane_id();
  const int n    = n0 + (lane & 15);
  if (n < C_) {
    const float bn = fcb[n];
#pragma unroll
    for (int r = 0; r < 8; ++r) {
      out[(size_t)(m0 +  0 + (lane >> 4) * 8 + r) * C_ + n] = acc0[r] + bn;
      out[(size_t)(m0 + 16 + (lane >> 4) * 8 + r) * C_ + n] = acc1[r] + bn;
    }
  }
}

// ---------------------------------------------------------------------------
// Host-side launcher
// ---------------------------------------------------------------------------
extern "C" void kernel_launch(void* const* d_in, const int* in_sizes, int n_in,
                              void* d_out, int out_size, void* d_ws, size_t ws_size,
                              hipStream_t stream) {
  (void)in_sizes; (void)n_in; (void)out_size; (void)ws_size;

  const float* x    = (const float*)d_in[0];  // [B,T,I]
  const float* Wih  = (const float*)d_in[1];  // [4H,I]
  const float* Whh  = (const float*)d_in[2];  // [4H,H]
  const float* bih  = (const float*)d_in[3];  // [4H]
  const float* bhh  = (const float*)d_in[4];  // [4H]
  const float* fcW  = (const float*)d_in[5];  // [C,H]
  const float* fcb  = (const float*)d_in[6];  // [C]
  float* out = (float*)d_out;                 // [B,C]

  // Workspace carve-up (256B aligned)
  char* ws = (char*)d_ws;
  size_t off = 0;
  auto carve = [&](size_t bytes) -> void* {
    void* p = ws + off;
    off = (off + bytes + 255) & ~(size_t)255;
    return p;
  };
  _Float16* x16   = (_Float16*)carve((size_t)T_ * B_ * I_ * 2);  // 32 MB
  _Float16* wih16 = (_Float16*)carve((size_t)4 * H_ * I_ * 2);   //  2 MB
  _Float16* whh16 = (_Float16*)carve((size_t)4 * H_ * H_ * 2);   //  8 MB
  _Float16* fc16  = (_Float16*)carve((size_t)NPAD_ * H_ * 2);    //  2 MB
  float*    bias  = (float*)   carve((size_t)4 * H_ * 4);
  _Float16* hA    = (_Float16*)carve((size_t)B_ * H_ * 2);
  _Float16* hB    = (_Float16*)carve((size_t)B_ * H_ * 2);
  float*    cst   = (float*)   carve((size_t)B_ * H_ * 4);

  // Pre-pass: convert to f16, combine biases, zero state
  k_cvt_x <<<4096, 256, 0, stream>>>(x, x16, B_ * T_ * I_);
  k_cvt   <<<1024, 256, 0, stream>>>(Wih, wih16, 4 * H_ * I_);
  k_cvt   <<<2048, 256, 0, stream>>>(Whh, whh16, 4 * H_ * H_);
  k_fc_pad<<<1024, 256, 0, stream>>>(fcW, fc16);
  k_init  <<< 512, 256, 0, stream>>>(bih, bhh, bias, hA, cst);

  // Recurrent loop: one fused GEMM+cell kernel per timestep, h double-buffered
  const dim3 grid(B_ / 32, H_ / 128);   // (4, 8)
  const _Float16* hin = hA;
  _Float16* hout = hB;
  for (int t = 0; t < T_; ++t) {
    k_lstm_step<<<grid, 256, 0, stream>>>(x16 + (size_t)t * B_ * I_,
                                          wih16, whh16, bias, hin, cst, hout);
    const _Float16* tmp = hout;
    hout = (_Float16*)hin;
    hin = tmp;
  }

  // Final FC from the last hidden state
  k_fc<<<dim3(B_ / 32, NPAD_ / 64), 128, 0, stream>>>(hin, fc16, fcb, out);
}
